// CaptchaRecognizer_59571196395858
// MI455X (gfx1250) — matrix operand order repaired and checked by reading, not compiled, not run on hardware
//
#include <hip/hip_runtime.h>
#include <hip/hip_bf16.h>

typedef __attribute__((ext_vector_type(16))) _Float16 v16h;
typedef __attribute__((ext_vector_type(8)))  _Float16 v8h;
typedef __attribute__((ext_vector_type(8)))  float    v8f;

#define A_MEM 0.1f      // DT * TAU_MEM_INV
#define D_SYN 0.8f      // 1 - DT * TAU_SYN_INV

// ---------------- elementwise init ----------------
__global__ void snn_zero_kernel(float* __restrict__ p, int n) {
  int i = blockIdx.x * blockDim.x + threadIdx.x;
  if (i < n) p[i] = 0.f;
}

__global__ void snn_fill_kernel(float* __restrict__ p, int n, float v) {
  int i = blockIdx.x * blockDim.x + threadIdx.x;
  if (i < n) p[i] = v;
}

// fp32 weight (out_f x in_f, row-major) -> f16 padded (N_pad x K_pad), zero pad
__global__ void snn_convert_w(const float* __restrict__ w, _Float16* __restrict__ wf,
                              int out_f, int in_f, int K_pad, int total) {
  int i = blockIdx.x * blockDim.x + threadIdx.x;
  if (i >= total) return;
  int n = i / K_pad;
  int k = i - n * K_pad;
  float v = (n < out_f && k < in_f) ? w[(size_t)n * in_f + k] : 0.f;
  wf[i] = (_Float16)v;
}

// ---------------- layer-0: constant-current LIF encoder + LIF cell ----------------
__global__ void snn_spike0(const float* __restrict__ x, float* __restrict__ enc_v,
                           float* __restrict__ lif_v, float* __restrict__ lif_i,
                           _Float16* __restrict__ z, int in_f, int K_pad) {
  int i = blockIdx.x * blockDim.x + threadIdx.x;
  int total = 128 * K_pad;
  if (i >= total) return;
  int b = i / K_pad;
  int f = i - b * K_pad;
  if (f < in_f) {
    size_t s = (size_t)b * in_f + f;
    // encoder: voltage-only LIF
    float ve = enc_v[s];
    ve += A_MEM * (x[s] - ve);
    float ze = (ve > 1.f) ? 1.f : 0.f;
    ve -= ze * ve;
    enc_v[s] = ve;
    // LIF cell (input = encoder spike)
    float v = lif_v[s], cur = lif_i[s];
    float vdec = v + A_MEM * (cur - v);
    float zs = (vdec > 1.f) ? 1.f : 0.f;
    lif_v[s] = (1.f - zs) * vdec;
    lif_i[s] = cur * D_SYN + ze;
    z[i] = (_Float16)zs;
  }
  // pad region (f >= in_f) stays zero from init; never written
}

// ---------------- WMMA GEMM (z @ W^T) + LI cell + fused next-layer LIF ----------------
// z : [128 x K_pad] f16 row-major (binary spikes, K zero-padded)
// wf: [N_pad x K_pad] f16 row-major
// Block: 256 thr = 8 waves = 4 m-pairs x 2 n-tiles; covers all M=128, N=32 cols.
// A tile (128 x 32 halves) double-buffered in LDS via async global->LDS DMA.
__global__ __launch_bounds__(256) void snn_gemm_li(
    const _Float16* __restrict__ z, const _Float16* __restrict__ wf,
    float* __restrict__ li_v, float* __restrict__ li_i,
    float* __restrict__ lifv_n, float* __restrict__ lifi_n,
    _Float16* __restrict__ z_next, int KPn,
    float* __restrict__ out_max,
    int K_pad, int out_f, int N_pad) {
  // LDS: 2 buffers x 128 rows x 40 halves (80B row stride -> conflict-free frags)
  __shared__ __align__(16) _Float16 smem[2 * 128 * 40];
  const int tid   = threadIdx.x;
  const int lane  = tid & 31;
  const int wave  = tid >> 5;
  const int lm    = lane & 15;
  const int g     = lane >> 4;
  const int mpair = wave & 3;                     // M tiles {2*mpair, 2*mpair+1}
  int ntile = blockIdx.x * 2 + (wave >> 2);
  const bool owner = (ntile * 16) < N_pad;
  if (!owner) ntile = (N_pad >> 4) - 1;           // clamp; stay in barriers
  const int n = ntile * 16 + lm;
  const _Float16* wr = wf + (size_t)n * K_pad;

  // async staging: thread t copies 32B of A row (t>>1), 32B-half (t&1)
  const int arow = tid >> 1;
  const int ahalf = tid & 1;
  const _Float16* gsrc = z + (size_t)arow * K_pad + ahalf * 16;
  const unsigned lds_tb =
      (unsigned)(unsigned long long)&smem[0] + (unsigned)(arow * 80 + ahalf * 32);

  v8f acc0 = {};
  v8f acc1 = {};
  const int S = K_pad >> 5;   // K chunks of 32

  // prologue: kick off chunk 0 into buffer 0
  {
    const _Float16* ga = gsrc;
    unsigned la = lds_tb;
    asm volatile(
        "global_load_async_to_lds_b128 %0, %1, off\n\t"
        "global_load_async_to_lds_b128 %0, %1, off offset:16"
        :: "v"(la), "v"(ga) : "memory");
  }

  for (int s = 0; s < S; ++s) {
    asm volatile("s_wait_asynccnt 0x0" ::: "memory");   // chunk s landed (this wave)
    __syncthreads();                                    // all landed + prior readers done
    if (s + 1 < S) {                                    // kick off chunk s+1 (other buf)
      const _Float16* ga = gsrc + (size_t)(s + 1) * 32;
      unsigned la = lds_tb + (unsigned)(((s + 1) & 1) * (128 * 80));
      asm volatile(
          "global_load_async_to_lds_b128 %0, %1, off\n\t"
          "global_load_async_to_lds_b128 %0, %1, off offset:16"
          :: "v"(la), "v"(ga) : "memory");
    }
    const int k0 = s << 5;
    v16h bfr = *(const v16h*)(wr + k0 + g * 16);        // B: 16 contiguous halves, col=lm
    if (k0 + 128 < K_pad)
      __builtin_prefetch((const void*)(wr + k0 + 128), 0, 3);
    const _Float16* sb = smem + (s & 1) * (128 * 40);
    {
      const _Float16* ap = sb + (mpair * 32 + lm) * 40 + g * 16;
      v8h lo = *(const v8h*)(ap);
      v8h hi = *(const v8h*)(ap + 8);
      v16h a = __builtin_shufflevector(lo, hi, 0,1,2,3,4,5,6,7,8,9,10,11,12,13,14,15);
      acc0 = __builtin_amdgcn_wmma_f32_16x16x32_f16(false, a, false, bfr, (short)0, acc0,
                                                    false, false);
    }
    {
      const _Float16* ap = sb + (mpair * 32 + 16 + lm) * 40 + g * 16;
      v8h lo = *(const v8h*)(ap);
      v8h hi = *(const v8h*)(ap + 8);
      v16h a = __builtin_shufflevector(lo, hi, 0,1,2,3,4,5,6,7,8,9,10,11,12,13,14,15);
      acc1 = __builtin_amdgcn_wmma_f32_16x16x32_f16(false, a, false, bfr, (short)0, acc1,
                                                    false, false);
    }
  }

  // Epilogue: LI cell update + fused next-layer LIF + spike emit (D row = r+8g, col = lm)
  if (owner && n < out_f) {
#define SNN_EPI(MRBASE, ACC)                                                     \
    _Pragma("unroll")                                                            \
    for (int r = 0; r < 8; ++r) {                                                \
      int mr = (MRBASE) + 8 * g + r;                                             \
      size_t idx = (size_t)mr * out_f + n;                                       \
      float ij = li_i[idx] + (ACC)[r];                                           \
      float vo = li_v[idx];                                                      \
      float vn = vo + A_MEM * (ij - vo);                                         \
      li_i[idx] = ij * D_SYN;                                                    \
      li_v[idx] = vn;                                                            \
      if (z_next) {                                                              \
        float lv = lifv_n[idx], lc = lifi_n[idx];                                \
        float vdec = lv + A_MEM * (lc - lv);                                     \
        float zs = (vdec > 1.f) ? 1.f : 0.f;                                     \
        lifv_n[idx] = (1.f - zs) * vdec;                                         \
        lifi_n[idx] = lc * D_SYN + vn;                                           \
        z_next[(size_t)mr * KPn + n] = (_Float16)zs;                             \
      } else {                                                                   \
        out_max[idx] = fmaxf(out_max[idx], vn);                                  \
      }                                                                          \
    }
    SNN_EPI(mpair * 32, acc0)
    SNN_EPI(mpair * 32 + 16, acc1)
#undef SNN_EPI
  }
}

// ---------------- final log_softmax over [128 x 10] ----------------
__global__ void snn_logsoftmax(const float* __restrict__ om, float* __restrict__ out) {
  int b = blockIdx.x * blockDim.x + threadIdx.x;
  if (b >= 128) return;
  float m = -INFINITY;
  for (int j = 0; j < 10; ++j) m = fmaxf(m, om[b * 10 + j]);
  float s = 0.f;
  for (int j = 0; j < 10; ++j) s += __expf(om[b * 10 + j] - m);
  float lse = __logf(s) + m;
  for (int j = 0; j < 10; ++j) out[b * 10 + j] = om[b * 10 + j] - lse;
}

extern "C" void kernel_launch(void* const* d_in, const int* in_sizes, int n_in,
                              void* d_out, int out_size, void* d_ws, size_t ws_size,
                              hipStream_t stream) {
  (void)in_sizes; (void)n_in; (void)out_size; (void)ws_size;
  const float* images = (const float*)d_in[0];
  const float* w[6];
  for (int k = 0; k < 6; ++k) w[k] = (const float*)d_in[1 + k];
  float* out = (float*)d_out;

  static const int IN_F[6]  = {12000, 2000, 1500, 1000, 500, 100};
  static const int OUT_F[6] = {2000, 1500, 1000, 500, 100, 10};
  static const int KP[6]    = {12000, 2016, 1504, 1024, 512, 128};   // K padded to x32
  static const int NP[6]    = {2000, 1504, 1008, 512, 112, 16};      // N padded to x16
  const int T = 32;

  // bump allocator over workspace (256B aligned; all sizes are 256B multiples)
  char* base = (char*)d_ws;
  size_t off = 0;
  auto alloc = [&](size_t bytes) -> char* {
    char* p = base + off;
    off = (off + bytes + 255) & ~(size_t)255;
    return p;
  };

  _Float16* wf16[6];
  for (int k = 0; k < 6; ++k)
    wf16[k] = (_Float16*)alloc((size_t)NP[k] * KP[k] * sizeof(_Float16));

  // per-layer spike buffers (contiguous -> one zero pass; pads stay 0 forever)
  _Float16* zb[6];
  for (int k = 0; k < 6; ++k)
    zb[k] = (_Float16*)alloc((size_t)128 * KP[k] * sizeof(_Float16));
  size_t zHalves = 0;
  for (int k = 0; k < 6; ++k) zHalves += (size_t)128 * KP[k];

  // contiguous state block
  float* enc_v = (float*)alloc((size_t)128 * 12000 * sizeof(float));
  float *lifv[6], *lifi[6], *liv[6], *lii[6];
  for (int k = 0; k < 6; ++k) lifv[k] = (float*)alloc((size_t)128 * IN_F[k] * sizeof(float));
  for (int k = 0; k < 6; ++k) lifi[k] = (float*)alloc((size_t)128 * IN_F[k] * sizeof(float));
  for (int k = 0; k < 6; ++k) liv[k]  = (float*)alloc((size_t)128 * OUT_F[k] * sizeof(float));
  for (int k = 0; k < 6; ++k) lii[k]  = (float*)alloc((size_t)128 * OUT_F[k] * sizeof(float));
  float* out_max = (float*)alloc((size_t)1280 * sizeof(float));

  // re-init every call (deterministic replay)
  int zeroF = (int)(zHalves / 2);                              // z region as floats
  snn_zero_kernel<<<(zeroF + 255) / 256, 256, 0, stream>>>((float*)zb[0], zeroF);
  int stateN = 128 * (12000 + 2 * 17100 + 2 * 5110);           // enc + lif(v,i) + li(v,i)
  snn_zero_kernel<<<(stateN + 255) / 256, 256, 0, stream>>>(enc_v, stateN);
  snn_fill_kernel<<<(1280 + 255) / 256, 256, 0, stream>>>(out_max, 1280, -INFINITY);

  // fp32 -> f16 weights (58MB total: L2-resident for all 32 timesteps)
  for (int k = 0; k < 6; ++k) {
    int tot = NP[k] * KP[k];
    snn_convert_w<<<(tot + 255) / 256, 256, 0, stream>>>(w[k], wf16[k], OUT_F[k], IN_F[k],
                                                         KP[k], tot);
  }

  for (int t = 0; t < T; ++t) {
    // encoder + layer-0 LIF -> z0
    int nA = 128 * KP[0];
    snn_spike0<<<(nA + 255) / 256, 256, 0, stream>>>(images, enc_v, lifv[0], lifi[0],
                                                     zb[0], IN_F[0], KP[0]);
    // each GEMM layer also performs the NEXT layer's LIF step and emits its spikes
    for (int k = 0; k < 6; ++k) {
      int gB = (NP[k] + 31) / 32;
      bool last = (k == 5);
      snn_gemm_li<<<gB, 256, 0, stream>>>(
          zb[k], wf16[k], liv[k], lii[k],
          last ? (float*)nullptr : lifv[k + 1],
          last ? (float*)nullptr : lifi[k + 1],
          last ? (_Float16*)nullptr : zb[k + 1],
          last ? 0 : KP[k + 1],
          last ? out_max : (float*)nullptr,
          KP[k], OUT_F[k], NP[k]);
    }
  }

  snn_logsoftmax<<<1, 128, 0, stream>>>(out_max, out);
}
